// HexGraphEncoder_43894565765174
// MI455X (gfx1250) — matrix operand order repaired and checked by reading, not compiled, not run on hardware
//
#include <hip/hip_runtime.h>
#include <hip/hip_bf16.h>

typedef __attribute__((ext_vector_type(16))) _Float16 v16h;
typedef __attribute__((ext_vector_type(8)))  float    v8f;

#define BB   32
#define NN   4096
#define EE   24576
#define HID  64
#define EMB  128
#define SLOPE 0.1f

// ---------------- zero fill ----------------
__global__ void hex_zero(unsigned int* __restrict__ p, long n) {
    long i = (long)blockIdx.x * blockDim.x + threadIdx.x;
    if (i < n) p[i] = 0u;
}

// ---------------- pre-swizzle W2s/W2n into WMMA B-fragment order ----------------
// Bfrag[gemm][kb][t][lane][h]  (8192 halves)
// B order used here (32x16, contiguous k): N = lane%16 ; K = (lane<16 ? h : 16+h) + 32*kb
__global__ void hex_prep_bfrag(const float* __restrict__ W2s,
                               const float* __restrict__ W2n,
                               _Float16* __restrict__ Bfrag) {
    int tid = blockIdx.x * blockDim.x + threadIdx.x;
    if (tid >= 2 * 2 * 4 * 32 * 16) return;
    int h    = tid & 15;
    int lane = (tid >> 4) & 31;
    int t    = (tid >> 9) & 3;
    int kb   = (tid >> 11) & 1;
    int gemm = (tid >> 12) & 1;
    int K    = ((lane & 16) ? 16 + h : h) + 32 * kb;
    int col  = (t << 4) + (lane & 15);
    const float* W = gemm ? W2n : W2s;
    Bfrag[tid] = (_Float16)W[K * HID + col];
}

// ---------------- layer-1 scalar scatter: S[b,n] = sum x[b,src], cnt[n] ----------------
__global__ void hex_edge_scalar(const float* __restrict__ node,
                                const int* __restrict__ src,
                                const int* __restrict__ dst,
                                float* __restrict__ S,
                                int* __restrict__ cnt) {
    int tid = blockIdx.x * blockDim.x + threadIdx.x;
    if (tid >= BB * EE) return;
    int b = tid / EE;
    int e = tid - b * EE;
    int s = src[e], d = dst[e];
    atomicAdd(&S[b * NN + d], node[b * NN + s]);
    if (b == 0) atomicAdd(&cnt[d], 1);
}

// ---------------- layer 1 (IN_DIM==1 -> elementwise), write x1 as f16 ----------------
__global__ void hex_layer1(const float* __restrict__ node,
                           const float* __restrict__ S,
                           const int* __restrict__ cnt,
                           const int* __restrict__ deg,
                           const float* __restrict__ W1s, const float* __restrict__ b1s,
                           const float* __restrict__ W1n, const float* __restrict__ b1n,
                           _Float16* __restrict__ x1h) {
    long tid = (long)blockIdx.x * blockDim.x + threadIdx.x;
    if (tid >= (long)BB * NN * HID) return;
    int h   = (int)(tid & 63);
    long idx = tid >> 6;               // b*NN + n
    int n   = (int)(idx & (NN - 1));
    float xv = node[idx];
    float sv = S[idx];
    float cv = (float)cnt[n];
    float df = (float)deg[n]; if (df < 1.f) df = 1.f;
    float v = xv * W1s[h] + b1s[h] + (sv * W1n[h] + cv * b1n[h]) / df;
    v = v > 0.f ? v : SLOPE * v;
    x1h[tid] = (_Float16)v;
}

// ---------------- layer-2 vector scatter: G[b,dst,:] += x1[b,src,:] ----------------
__global__ void hex_scatter_vec(const _Float16* __restrict__ x1h,
                                const int* __restrict__ src,
                                const int* __restrict__ dst,
                                float* __restrict__ G) {
    int tid = blockIdx.x * blockDim.x + threadIdx.x;
    if (tid >= BB * EE) return;
    int b = tid / EE;
    int e = tid - b * EE;
    const _Float16* xs = x1h + ((long)b * NN + src[e]) * HID;
    float* g = G + ((long)b * NN + dst[e]) * HID;
    #pragma unroll 8
    for (int h = 0; h < HID; ++h) atomicAdd(&g[h], (float)xs[h]);
}

// ---------------- layer 2: fused dual WMMA GEMM + epilogue + mean-pool reduce ----------------
// One wave handles 16 consecutive rows of the flattened (b,n) space.
__global__ __launch_bounds__(128) void hex_layer2_wmma(
    const _Float16* __restrict__ x1h, const float* __restrict__ G,
    const _Float16* __restrict__ Bfrag,
    const float* __restrict__ b2s, const float* __restrict__ b2n,
    const int* __restrict__ cnt, const int* __restrict__ deg,
    float* __restrict__ pooled) {
    const int lane   = threadIdx.x & 31;
    const int wid    = blockIdx.x * (blockDim.x >> 5) + (threadIdx.x >> 5);
    const int rowStart = wid << 4;                 // 16 rows per wave
    const int b      = rowStart >> 12;             // / NN
    const int n0     = rowStart & (NN - 1);
    const int laneLo = lane & 15;
    const int hiOff  = (lane >> 4) << 3;           // 0 or 8

    // ---- A fragments (16x32, contiguous k order used here: K(h) = h + 2*hiOff) ----
    const long aBase = ((long)b * NN + n0 + laneLo) * HID;
    const _Float16* xr = x1h + aBase;
    const float*    gr = G   + aBase;
    v16h aS0, aS1, aN0, aN1;
    #pragma unroll
    for (int h = 0; h < 16; ++h) {
        int k = h + (hiOff << 1);
        aS0[h] = xr[k];
        aS1[h] = xr[k + 32];
        aN0[h] = (_Float16)gr[k];
        aN1[h] = (_Float16)gr[k + 32];
    }

    // per-row scale factors (C/D layout: VGPR r <-> row r + hiOff)
    float invd[8], cntf[8];
    #pragma unroll
    for (int r = 0; r < 8; ++r) {
        int n = n0 + r + hiOff;
        float df = (float)deg[n]; if (df < 1.f) df = 1.f;
        invd[r] = 1.f / df;
        cntf[r] = (float)cnt[n];
    }

    const v16h* bf = (const v16h*)Bfrag;

    #pragma unroll
    for (int t = 0; t < 4; ++t) {
        v16h b00 = bf[((0 * 4 + t) * 32) + lane];  // W2s, kb=0
        v16h b01 = bf[((1 * 4 + t) * 32) + lane];  // W2s, kb=1
        v16h b10 = bf[((2 * 4 + t) * 32) + lane];  // W2n, kb=0
        v16h b11 = bf[((3 * 4 + t) * 32) + lane];  // W2n, kb=1
        v8f cs = {}, cn = {};
        cs = __builtin_amdgcn_wmma_f32_16x16x32_f16(false, aS0, false, b00, (short)0, cs, false, false);
        cs = __builtin_amdgcn_wmma_f32_16x16x32_f16(false, aS1, false, b01, (short)0, cs, false, false);
        cn = __builtin_amdgcn_wmma_f32_16x16x32_f16(false, aN0, false, b10, (short)0, cn, false, false);
        cn = __builtin_amdgcn_wmma_f32_16x16x32_f16(false, aN1, false, b11, (short)0, cn, false, false);

        int col = (t << 4) + laneLo;
        float bs = b2s[col], bn = b2n[col];
        float psum = 0.f;
        #pragma unroll
        for (int r = 0; r < 8; ++r) {
            float v = cs[r] + bs + (cn[r] + cntf[r] * bn) * invd[r];
            v = v > 0.f ? v : SLOPE * v;
            psum += v;
        }
        atomicAdd(&pooled[(b << 6) + col], psum);
    }
}

// ---------------- head MLP: pooled/N -> lrelu(@Wp1+b) -> @Wp2+b ----------------
__global__ __launch_bounds__(256) void hex_head(const float* __restrict__ pooled,
                                                const float* __restrict__ Wp1, const float* __restrict__ bp1,
                                                const float* __restrict__ Wp2, const float* __restrict__ bp2,
                                                float* __restrict__ out) {
    __shared__ float hbuf[BB * HID];
    const float invN = 1.0f / (float)NN;
    for (int i = threadIdx.x; i < BB * HID; i += blockDim.x) {
        int r = i >> 6, c = i & 63;
        float acc = bp1[c];
        for (int k = 0; k < HID; ++k)
            acc += pooled[r * HID + k] * invN * Wp1[k * HID + c];
        hbuf[i] = acc > 0.f ? acc : SLOPE * acc;
    }
    __syncthreads();
    for (int i = threadIdx.x; i < BB * EMB; i += blockDim.x) {
        int r = i >> 7, c = i & 127;
        float acc = bp2[c];
        for (int k = 0; k < HID; ++k)
            acc += hbuf[r * HID + k] * Wp2[k * EMB + c];
        out[i] = acc;
    }
}

extern "C" void kernel_launch(void* const* d_in, const int* in_sizes, int n_in,
                              void* d_out, int out_size, void* d_ws, size_t ws_size,
                              hipStream_t stream) {
    const float* node = (const float*)d_in[0];
    const int*   eidx = (const int*)d_in[1];
    const int*   deg  = (const int*)d_in[2];
    const float* W1s  = (const float*)d_in[3];
    const float* b1s  = (const float*)d_in[4];
    const float* W1n  = (const float*)d_in[5];
    const float* b1n  = (const float*)d_in[6];
    const float* W2s  = (const float*)d_in[7];
    const float* b2s  = (const float*)d_in[8];
    const float* W2n  = (const float*)d_in[9];
    const float* b2n  = (const float*)d_in[10];
    const float* Wp1  = (const float*)d_in[11];
    const float* bp1  = (const float*)d_in[12];
    const float* Wp2  = (const float*)d_in[13];
    const float* bp2  = (const float*)d_in[14];
    const int* srcIdx = eidx;
    const int* dstIdx = eidx + EE;

    // workspace layout (zeroed region first, contiguous)
    char* ws = (char*)d_ws;
    float*    G      = (float*)(ws);                            // 32 MB
    float*    S      = (float*)(ws + 33554432);                 // 512 KB
    int*      cnt    = (int*)  (ws + 34078720);                 // 16 KB
    float*    pooled = (float*)(ws + 34095104);                 // 8 KB
    _Float16* x1h    = (_Float16*)(ws + 34103296);              // 16 MB
    _Float16* Bfrag  = (_Float16*)(ws + 50880512);              // 16 KB

    const long zeroWords = (long)BB * NN * HID + (long)BB * NN + NN + BB * HID;
    hex_zero<<<(unsigned)((zeroWords + 255) / 256), 256, 0, stream>>>((unsigned int*)ws, zeroWords);

    hex_prep_bfrag<<<32, 256, 0, stream>>>(W2s, W2n, Bfrag);

    hex_edge_scalar<<<(BB * EE + 255) / 256, 256, 0, stream>>>(node, srcIdx, dstIdx, S, cnt);

    hex_layer1<<<(unsigned)(((long)BB * NN * HID + 255) / 256), 256, 0, stream>>>(
        node, S, cnt, deg, W1s, b1s, W1n, b1n, x1h);

    hex_scatter_vec<<<(BB * EE + 255) / 256, 256, 0, stream>>>(x1h, srcIdx, dstIdx, G);

    // 8192 waves of 16 rows each, 4 waves per block
    hex_layer2_wmma<<<2048, 128, 0, stream>>>(x1h, G, Bfrag, b2s, b2n, cnt, deg, pooled);

    hex_head<<<1, 256, 0, stream>>>(pooled, Wp1, bp1, Wp2, bp2, (float*)d_out);
}
